// RecurrentRetention_3281355014306
// MI455X (gfx1250) — matrix operand order repaired and checked by reading, not compiled, not run on hardware
//
#include <hip/hip_runtime.h>
#include <hip/hip_bf16.h>

#define GAMMA 0.9f

typedef __attribute__((ext_vector_type(16))) _Float16 v16h;
typedef __attribute__((ext_vector_type(8)))  float    v8f;

static constexpr int Bsz = 16;
static constexpr int Tsz = 2048;
static constexpr int Dsz = 256;

// ---------------------------------------------------------------------------
// wvsum[k] = sum_j Wv[k, j]   (Wv row-sums; reduces V GEMM to a matvec)
// ---------------------------------------------------------------------------
__global__ __launch_bounds__(256) void wv_rowsum_kernel(const float* __restrict__ Wv,
                                                        float* __restrict__ wvsum) {
    int k = threadIdx.x;                 // 0..255
    const float* row = Wv + k * Dsz;
    float s = 0.f;
    #pragma unroll 8
    for (int j = 0; j < Dsz; ++j) s += row[j];
    wvsum[k] = s;
}

// ---------------------------------------------------------------------------
// x (f32) -> xh (f16), elementwise
// ---------------------------------------------------------------------------
__global__ void xh_convert_kernel(const float* __restrict__ x,
                                  _Float16* __restrict__ xh, int n) {
    int i = blockIdx.x * blockDim.x + threadIdx.x;
    if (i < n) xh[i] = (_Float16)x[i];
}

// ---------------------------------------------------------------------------
// WqT[n*256+k] = (f16)Wq[k*256+n]; same for Wk. Transposed so a WMMA B-column
// fragment (16 consecutive K at fixed n) is contiguous in memory.
// ---------------------------------------------------------------------------
__global__ void wT_convert_kernel(const float* __restrict__ Wq,
                                  const float* __restrict__ Wk,
                                  _Float16* __restrict__ WqT,
                                  _Float16* __restrict__ WkT) {
    int i = blockIdx.x * blockDim.x + threadIdx.x;   // i = n*256 + k
    if (i >= Dsz * Dsz) return;
    int n = i >> 8;
    int k = i & 255;
    WqT[i] = (_Float16)Wq[k * Dsz + n];
    WkT[i] = (_Float16)Wk[k * Dsz + n];
}

// ---------------------------------------------------------------------------
// vsum[row] = x[row,:] . wvsum   (one wave per row, f32 accumulation)
// ---------------------------------------------------------------------------
__global__ __launch_bounds__(32) void vsum_kernel(const float* __restrict__ x,
                                                  const float* __restrict__ wvsum,
                                                  float* __restrict__ vsum) {
    int row  = blockIdx.x;
    int lane = threadIdx.x;
    const float* xr = x + (size_t)row * Dsz;
    float s = 0.f;
    #pragma unroll
    for (int j = 0; j < Dsz / 32; ++j)
        s += xr[lane + j * 32] * wvsum[lane + j * 32];
    #pragma unroll
    for (int off = 16; off > 0; off >>= 1)
        s += __shfl_xor(s, off, 32);
    if (lane == 0) vsum[row] = s;
}

// ---------------------------------------------------------------------------
// Fused Q/K GEMM + batch-reduced c epilogue.
// One wave per (16-t x 16-n) output tile; loops all 16 batches so that
// c[t][n] = sum_b K[b,t,n]*vsum[b,t] accumulates in registers.
// Q is written straight into d_out (scaled in-place later).
// Uses v_wmma_f32_16x16x32_f16 (f16 in, f32 accum).
// ---------------------------------------------------------------------------
__global__ __launch_bounds__(32) void retention_gemm_kernel(
    const _Float16* __restrict__ xh,    // [B*T, 256] f16
    const _Float16* __restrict__ WqT,   // [256, 256] f16 (n-major)
    const _Float16* __restrict__ WkT,   // [256, 256] f16 (n-major)
    const float*    __restrict__ vsum,  // [B*T]
    float*          __restrict__ Qout,  // [B*T, 256]  (== d_out)
    float*          __restrict__ cOut)  // [T, 256]
{
    const int ttile = blockIdx.x * 16;   // 128 t-tiles
    const int ntile = blockIdx.y * 16;   // 16 n-tiles
    const int lane  = threadIdx.x;
    const int lmod  = lane & 15;
    const int lhi   = lane >> 4;         // 0 or 1

    // --- B fragments, both weight matrices, all 8 K-steps, loaded once ---
    // B 32x16 f16 k order used here (contiguous, same as A): lanes 0-15 hold K=0..15 (col = lane),
    // lanes 16-31 hold K=16..31 (col = lane-16); packed f16 pairs per VGPR.
    v16h bq[8], bk[8];
    {
        const int ncol = ntile + lmod;
        #pragma unroll
        for (int kk = 0; kk < 8; ++kk) {
            int off = ncol * Dsz + kk * 32 + lhi * 16;
            union { uint4 u[2]; v16h h; } tq, tk;
            tq.u[0] = *(const uint4*)(WqT + off);
            tq.u[1] = *(const uint4*)(WqT + off + 8);
            tk.u[0] = *(const uint4*)(WkT + off);
            tk.u[1] = *(const uint4*)(WkT + off + 8);
            bq[kk] = tq.h;
            bk[kk] = tk.h;
        }
    }

    v8f cacc = {};
    #pragma unroll 1
    for (int b = 0; b < Bsz; ++b) {
        // A 16x32 f16 k order used here (contiguous, same as B): lane row M = lane%16;
        // lanes 0-15: elems 0-15 = K0..15
        // lanes16-31: elems 0-15 = K16..31
        const _Float16* abase = xh + (size_t)(b * Tsz + ttile + lmod) * Dsz;
        v8f qacc = {}, kacc = {};
        #pragma unroll
        for (int kk = 0; kk < 8; ++kk) {
            int k0 = kk * 32 + lhi * 16;
            union { uint4 u[2]; v16h h; } ta;
            ta.u[0] = *(const uint4*)(abase + k0);
            ta.u[1] = *(const uint4*)(abase + k0 + 8);
            qacc = __builtin_amdgcn_wmma_f32_16x16x32_f16(
                false, ta.h, false, bq[kk], (short)0, qacc, false, false);
            kacc = __builtin_amdgcn_wmma_f32_16x16x32_f16(
                false, ta.h, false, bk[kk], (short)0, kacc, false, false);
        }
        // C/D layout: lane n = lane%16; VGPR g holds M = g + (lane>=16)*8
        #pragma unroll
        for (int g = 0; g < 8; ++g) {
            int m   = g + lhi * 8;
            int row = b * Tsz + ttile + m;
            Qout[(size_t)row * Dsz + ntile + lmod] = qacc[g];
            cacc[g] += kacc[g] * vsum[row];
        }
    }

    #pragma unroll
    for (int g = 0; g < 8; ++g) {
        int m = g + lhi * 8;
        cOut[(size_t)(ttile + m) * Dsz + ntile + lmod] = cacc[g];
    }
}

// ---------------------------------------------------------------------------
// Decay scan: rows[t] = GAMMA*rows[t-1] + c[t], t = 1..T-1; S[0] := S[1].
// 256 threads, one per channel d; loads/stores are fully coalesced.
// ---------------------------------------------------------------------------
__global__ __launch_bounds__(256) void scan_kernel(const float* __restrict__ c,
                                                   float* __restrict__ S) {
    int d = threadIdx.x;
    float r = c[1 * Dsz + d];         // t=1: gamma*0 + c[1]
    S[0 * Dsz + d] = r;               // S[0] duplicates the t=1 row
    S[1 * Dsz + d] = r;
    for (int t = 2; t < Tsz; ++t) {
        r = GAMMA * r + c[t * Dsz + d];
        S[t * Dsz + d] = r;
    }
}

// ---------------------------------------------------------------------------
// out[b,t,d] = Q[b,t,d] * S2[t,d], with S2[t,d] = Sflat[d*T + t]
// (the reference's reshape(D,T).T scramble). In-place on d_out.
// ---------------------------------------------------------------------------
__global__ void finalize_kernel(float* __restrict__ out,
                                const float* __restrict__ S, int n) {
    int i = blockIdx.x * blockDim.x + threadIdx.x;
    if (i >= n) return;
    int d = i & (Dsz - 1);
    int t = (i >> 8) & (Tsz - 1);
    out[i] *= S[d * Tsz + t];
}

// ---------------------------------------------------------------------------
extern "C" void kernel_launch(void* const* d_in, const int* in_sizes, int n_in,
                              void* d_out, int out_size, void* d_ws, size_t ws_size,
                              hipStream_t stream) {
    const float* x  = (const float*)d_in[0];   // [16, 2048, 256]
    const float* Wq = (const float*)d_in[1];   // [256, 256]
    const float* Wk = (const float*)d_in[2];
    const float* Wv = (const float*)d_in[3];
    float* out = (float*)d_out;                // [16, 2048, 256]

    const int BT  = Bsz * Tsz;                 // 32768
    const int BTD = BT * Dsz;                  // 8388608

    // Workspace layout (bytes, 256-aligned sections)
    char* ws = (char*)d_ws;
    size_t off = 0;
    _Float16* xh    = (_Float16*)(ws + off); off += (size_t)BTD * 2;        // 16 MB
    _Float16* WqT   = (_Float16*)(ws + off); off += (size_t)Dsz * Dsz * 2;  // 128 KB
    _Float16* WkT   = (_Float16*)(ws + off); off += (size_t)Dsz * Dsz * 2;  // 128 KB
    float*    wvsum = (float*)(ws + off);    off += (size_t)Dsz * 4;        // 1 KB
    float*    vsumA = (float*)(ws + off);    off += (size_t)BT * 4;         // 128 KB
    float*    cArr  = (float*)(ws + off);    off += (size_t)Tsz * Dsz * 4;  // 2 MB
    float*    SArr  = (float*)(ws + off);    off += (size_t)Tsz * Dsz * 4;  // 2 MB
    (void)ws_size; (void)in_sizes; (void)n_in; (void)out_size;

    wv_rowsum_kernel<<<1, 256, 0, stream>>>(Wv, wvsum);
    xh_convert_kernel<<<(BTD + 255) / 256, 256, 0, stream>>>(x, xh, BTD);
    wT_convert_kernel<<<(Dsz * Dsz + 255) / 256, 256, 0, stream>>>(Wq, Wk, WqT, WkT);
    vsum_kernel<<<BT, 32, 0, stream>>>(x, wvsum, vsumA);

    dim3 ggrid(Tsz / 16, Dsz / 16);            // 128 x 16 tiles, 1 wave each
    retention_gemm_kernel<<<ggrid, 32, 0, stream>>>(xh, WqT, WkT, vsumA, out, cArr);

    scan_kernel<<<1, 256, 0, stream>>>(cArr, SArr);
    finalize_kernel<<<(BTD + 255) / 256, 256, 0, stream>>>(out, SArr, BTD);
}
